// MyData_SNN_38199439130695
// MI455X (gfx1250) — matrix operand order repaired and checked by reading, not compiled, not run on hardware
//
#include <hip/hip_runtime.h>
#include <math.h>

typedef _Float16 half_t;
typedef __attribute__((ext_vector_type(8)))  _Float16 v8h;
typedef __attribute__((ext_vector_type(16))) _Float16 v16h;
typedef __attribute__((ext_vector_type(8)))  float    v8f;

// ---------------- problem dimensions ----------------
static constexpr int BB   = 32;          // batch
static constexpr int TT   = 64;          // time steps
static constexpr int BT   = BB * TT;     // 2048 frames
static constexpr int HID  = 128;
static constexpr int NCLS = 10;
static constexpr int FDIM = 96 * 12 * 12; // 13824

// LIF constants: DT*TAU_MEM_INV = 0.1, DT*TAU_SYN_INV = 0.2
static constexpr float KMEM = 0.1f;
static constexpr float KSYN = 0.8f;   // 1 - DT*TAU_SYN_INV
static constexpr float VTH  = 0.4f;

// ---------------- WMMA helpers (gfx1250 wave32) ----------------
// A fragment (16x32 f16): lane element e -> K = e + (lane>=16?16:0)
// of row M = lane%16. Two contiguous 16B runs per lane.
__device__ __forceinline__ v16h load_a_row(const half_t* __restrict__ rowp, int hi) {
  v8h lo = *(const v8h*)(rowp + hi * 16);
  v8h hh = *(const v8h*)(rowp + hi * 16 + 8);
  v16h a;
#pragma unroll
  for (int e = 0; e < 8; ++e) { a[e] = lo[e]; a[e + 8] = hh[e]; }
  return a;
}

__device__ __forceinline__ v8f wmma32f16(v16h a, v16h b, v8f c) {
  // (neg_a, A, neg_b, B, c_mod, C, reuse_a, reuse_b)
  return __builtin_amdgcn_wmma_f32_16x16x32_f16(false, a, false, b, (short)0, c,
                                                false, false);
}

// ---------------- weight fragment packing ----------------
// Packed B layout: pack[(kstep*NT + ntile)*32 + lane][e], e=0..15 contiguous;
// element e of lane corresponds to B[kstep*32 + (lane>=16?16:0) + e][ntile*16 + lane%16].

__global__ void __launch_bounds__(256)
pack_w2_k(const float* __restrict__ w2, half_t* __restrict__ pack2) {
  int idx = blockIdx.x * 256 + threadIdx.x;
  if (idx >= 9 * 4 * 32 * 16) return;
  int e    = idx & 15;
  int lane = (idx >> 4) & 31;
  int nt   = (idx >> 9) & 3;
  int ks   = idx >> 11;            // kh*3+kw
  int kh = ks / 3, kw = ks % 3;
  int cc = (lane >> 4) * 16 + e;   // input channel 0..31
  int n  = nt * 16 + (lane & 15);  // output channel 0..63
  pack2[idx] = (half_t)w2[((n * 32 + cc) * 3 + kh) * 3 + kw];
}

__global__ void __launch_bounds__(256)
pack_w3_k(const float* __restrict__ w3, half_t* __restrict__ pack3) {
  int idx = blockIdx.x * 256 + threadIdx.x;
  if (idx >= 18 * 6 * 32 * 16) return;
  int e    = idx & 15;
  int lane = (idx >> 4) & 31;
  int nt   = (idx >> 9) % 6;
  int s    = idx / 3072;           // (kh*3+kw)*2 + chunk
  int chunk = s & 1, kk = s >> 1;
  int kh = kk / 3, kw = kk % 3;
  int cc = chunk * 32 + (lane >> 4) * 16 + e;  // 0..63
  int n  = nt * 16 + (lane & 15);              // 0..95
  pack3[idx] = (half_t)w3[((n * 64 + cc) * 3 + kh) * 3 + kw];
}

// feats stored pixel-major: k = p*96 + c ; reference flatten is c*144 + p
__global__ void __launch_bounds__(256)
pack_win_k(const float* __restrict__ w_in, half_t* __restrict__ packin) {
  int idx = blockIdx.x * 256 + threadIdx.x;
  if (idx >= 432 * 8 * 32 * 16) return;
  int e    = idx & 15;
  int lane = (idx >> 4) & 31;
  int nt   = (idx >> 9) & 7;
  int s    = idx >> 12;                         // 0..431
  int klocal = (lane >> 4) * 16 + e;            // 0..31
  int cc = (s % 3) * 32 + klocal;               // channel 0..95
  int p  = s / 3;                               // pixel 0..143
  int h  = nt * 16 + (lane & 15);               // 0..127
  packin[idx] = (half_t)w_in[(size_t)h * FDIM + cc * 144 + p];
}

// recurrent: B[k][h] = w_rec[h][k]
__global__ void __launch_bounds__(256)
pack_wrec_k(const float* __restrict__ w_rec, half_t* __restrict__ packrec) {
  int idx = blockIdx.x * 256 + threadIdx.x;
  if (idx >= 4 * 8 * 32 * 16) return;
  int e    = idx & 15;
  int lane = (idx >> 4) & 31;
  int nt   = (idx >> 9) & 7;
  int s    = idx >> 12;                         // 0..3
  int k = s * 32 + (lane >> 4) * 16 + e;
  int h = nt * 16 + (lane & 15);
  packrec[idx] = (half_t)w_rec[h * HID + k];
}

// ---------------- conv1 (C_in=1, direct, VALU) ----------------
__global__ void __launch_bounds__(256)
conv1_k(const float* __restrict__ x, const float* __restrict__ w1,
        const float* __restrict__ b1, half_t* __restrict__ act1) {
  int idx = blockIdx.x * 256 + threadIdx.x;
  if (idx >= BT * 30 * 30 * 32) return;
  int oc = idx & 31;
  int t  = idx >> 5;
  int ox = t % 30; t /= 30;
  int oy = t % 30; t /= 30;
  int n  = t;
  const float* xp = x + (size_t)n * 4096 + (oy * 2) * 64 + ox * 2;
  const float* wp = w1 + oc * 25;
  float acc = b1[oc];
#pragma unroll
  for (int ky = 0; ky < 5; ++ky)
#pragma unroll
    for (int kx = 0; kx < 5; ++kx)
      acc = fmaf(xp[ky * 64 + kx], wp[ky * 5 + kx], acc);
  act1[idx] = (half_t)fmaxf(acc, 0.f);   // NHWC [BT][30][30][32]
}

// ---------------- conv2: implicit GEMM, M=16 pixels, N=64, K=288 ----------------
__global__ void __launch_bounds__(128)
conv2_wmma(const half_t* __restrict__ act1, const half_t* __restrict__ pack2,
           const float* __restrict__ b2, half_t* __restrict__ act2) {
  int blk   = blockIdx.x;
  int ptile = blk % 49;
  int frame = blk / 49;
  int lane  = threadIdx.x & 31;
  int ntile = threadIdx.x >> 5;     // 0..3
  int hi    = lane >> 4;
  int p  = ptile * 16 + (lane & 15);
  int oy = p / 28, ox = p % 28;
  const half_t* base = act1 + (size_t)frame * 30 * 30 * 32;
  v8f c = {};
#pragma unroll
  for (int kh = 0; kh < 3; ++kh)
#pragma unroll
    for (int kw = 0; kw < 3; ++kw) {
      const half_t* rowp = base + ((oy + kh) * 30 + (ox + kw)) * 32;
      v16h a = load_a_row(rowp, hi);
      int ks = kh * 3 + kw;
      v16h b = *(const v16h*)(pack2 + ((ks * 4 + ntile) * 32 + lane) * 16);
      c = wmma32f16(a, b, c);
    }
  int oc = ntile * 16 + (lane & 15);
  float bias = b2[oc];
  int pm = ptile * 16 + hi * 8;
#pragma unroll
  for (int r = 0; r < 8; ++r) {
    int pp = pm + r;
    int oyy = pp / 28, oxx = pp % 28;
    float val = fmaxf(c[r] + bias, 0.f);
    act2[((size_t)frame * 784 + oyy * 28 + oxx) * 64 + oc] = (half_t)val;
  }
}

// ---------------- 2x2 maxpool ----------------
__global__ void __launch_bounds__(256)
maxpool_k(const half_t* __restrict__ act2, half_t* __restrict__ act2p) {
  int idx = blockIdx.x * 256 + threadIdx.x;
  if (idx >= BT * 14 * 14 * 64) return;
  int oc = idx & 63;
  int t  = idx >> 6;
  int ox = t % 14; t /= 14;
  int oy = t % 14; t /= 14;
  int n  = t;
  const half_t* p = act2 + (((size_t)n * 28 + oy * 2) * 28 + ox * 2) * 64 + oc;
  float a = (float)p[0], b = (float)p[64];
  float cc = (float)p[28 * 64], d = (float)p[28 * 64 + 64];
  act2p[idx] = (half_t)fmaxf(fmaxf(a, b), fmaxf(cc, d));
}

// ---------------- conv3: implicit GEMM, M=16 pixels, N=96, K=576 ----------------
__global__ void __launch_bounds__(192)
conv3_wmma(const half_t* __restrict__ act2p, const half_t* __restrict__ pack3,
           const float* __restrict__ b3, half_t* __restrict__ feats) {
  int blk   = blockIdx.x;
  int ptile = blk % 9;
  int frame = blk / 9;
  int lane  = threadIdx.x & 31;
  int ntile = threadIdx.x >> 5;     // 0..5
  int hi    = lane >> 4;
  int p  = ptile * 16 + (lane & 15);
  int oy = p / 12, ox = p % 12;
  const half_t* base = act2p + (size_t)frame * 196 * 64;
  v8f c = {};
#pragma unroll
  for (int kh = 0; kh < 3; ++kh)
#pragma unroll
    for (int kw = 0; kw < 3; ++kw)
#pragma unroll
      for (int chunk = 0; chunk < 2; ++chunk) {
        const half_t* rowp = base + ((oy + kh) * 14 + (ox + kw)) * 64 + chunk * 32;
        v16h a = load_a_row(rowp, hi);
        int s = (kh * 3 + kw) * 2 + chunk;
        v16h b = *(const v16h*)(pack3 + ((s * 6 + ntile) * 32 + lane) * 16);
        c = wmma32f16(a, b, c);
      }
  int oc = ntile * 16 + (lane & 15);
  float bias = b3[oc];
  int pm = ptile * 16 + hi * 8;
#pragma unroll
  for (int r = 0; r < 8; ++r) {
    int pp = pm + r;
    float val = fmaxf(c[r] + bias, 0.f);
    feats[(size_t)frame * FDIM + pp * 96 + oc] = (half_t)val;  // pixel-major
  }
}

// ---------------- ff = feats @ w_in^T : M=2048, N=128, K=13824 ----------------
__global__ void __launch_bounds__(256)
ff_gemm(const half_t* __restrict__ feats, const half_t* __restrict__ packin,
        float* __restrict__ ff) {
  int mtile = blockIdx.x;           // 0..127
  int lane  = threadIdx.x & 31;
  int ntile = threadIdx.x >> 5;     // 0..7
  int hi    = lane >> 4;
  const half_t* arow = feats + (size_t)(mtile * 16 + (lane & 15)) * FDIM;
  v8f c = {};
  for (int s = 0; s < 432; ++s) {
    if (s + 8 < 432) __builtin_prefetch(arow + (s + 8) * 32, 0, 1);
    v16h a = load_a_row(arow + s * 32, hi);
    v16h b = *(const v16h*)(packin + ((size_t)(s * 8 + ntile) * 32 + lane) * 16);
    c = wmma32f16(a, b, c);
  }
  int m0 = mtile * 16 + hi * 8;
  int h  = ntile * 16 + (lane & 15);
#pragma unroll
  for (int r = 0; r < 8; ++r) ff[(size_t)(m0 + r) * HID + h] = c[r];
}

// ---------------- persistent recurrent LIF scan (1 block, 16 waves) ----------------
__global__ void __launch_bounds__(512)
snn_recurrent(const float* __restrict__ ff,        // indexed [(b*TT+t)*HID + h]
              const half_t* __restrict__ packrec,  // [4*8*32*16]
              const float* __restrict__ w_fc,      // [10][128]
              const float* __restrict__ b_fc,      // [10]
              float* __restrict__ out)             // [32][10]
{
  __shared__ half_t zs[BB * HID];          // spikes, f16
  __shared__ half_t wrec[4 * 8 * 32 * 16]; // packed w_rec^T fragments
  __shared__ float  vr[BB * NCLS], ir[BB * NCLS], vmax[BB * NCLS];

  int tid   = threadIdx.x;
  int lane  = tid & 31;
  int wv    = tid >> 5;      // 0..15
  int mtile = wv >> 3;       // 0..1
  int ntile = wv & 7;        // 0..7
  int hi    = lane >> 4;

  for (int i = tid; i < BB * HID; i += 512) zs[i] = (half_t)0.f;
  for (int i = tid; i < 4 * 8 * 32 * 16; i += 512) wrec[i] = packrec[i];
  if (tid < BB * NCLS) { vr[tid] = 0.f; ir[tid] = 0.f; vmax[tid] = -3.4e38f; }
  __syncthreads();

  float v[8], cur[8];
#pragma unroll
  for (int r = 0; r < 8; ++r) { v[r] = 0.f; cur[r] = 0.f; }

  int m0   = mtile * 16 + hi * 8;          // batch rows of the C-frag
  int hcol = ntile * 16 + (lane & 15);     // hidden column
  int arow = mtile * 16 + (lane & 15);     // batch row for A-frag addressing

  for (int t = 0; t < TT; ++t) {
    // rec = z_{t-1} @ w_rec^T  (2x8 tiles across 16 waves, K=128 -> 4 WMMAs)
    v8f c = {};
#pragma unroll
    for (int s = 0; s < 4; ++s) {
      v16h a = load_a_row(&zs[arow * HID + s * 32], hi);
      v16h b = *(const v16h*)(&wrec[((s * 8 + ntile) * 32 + lane) * 16]);
      c = wmma32f16(a, b, c);
    }
    __syncthreads();  // all reads of old zs complete

    half_t znew[8];
#pragma unroll
    for (int r = 0; r < 8; ++r) {
      int m = m0 + r;                              // batch index
      float vdec = v[r] + KMEM * (cur[r] - v[r]);
      float idec = KSYN * cur[r];
      float z = (vdec - VTH) > 0.f ? 1.f : 0.f;    // heaviside forward
      v[r] = (1.f - z) * vdec;
      float fft = ff[(size_t)(m * TT + t) * HID + hcol];
      cur[r] = idec + fft + c[r];
      znew[r] = (half_t)z;
    }
#pragma unroll
    for (int r = 0; r < 8; ++r) zs[(m0 + r) * HID + hcol] = znew[r];
    __syncthreads();  // new spikes visible

    // readout linear + LI cell + running max (320 lanes)
    if (tid < BB * NCLS) {
      int b = tid / NCLS, cc = tid % NCLS;
      float acc = b_fc[cc];
      for (int h = 0; h < HID; ++h)
        acc = fmaf((float)zs[b * HID + h], w_fc[cc * HID + h], acc);
      float vrn = vr[tid] + KMEM * (ir[tid] - vr[tid]);
      float irn = KSYN * ir[tid] + acc;
      vr[tid] = vrn; ir[tid] = irn;
      vmax[tid] = fmaxf(vmax[tid], vrn);
    }
    __syncthreads();
  }

  // log_softmax over classes
  if (tid < BB * NCLS) {
    int b = tid / NCLS;
    float mx = -3.4e38f;
    for (int j = 0; j < NCLS; ++j) mx = fmaxf(mx, vmax[b * NCLS + j]);
    float s = 0.f;
    for (int j = 0; j < NCLS; ++j) s += expf(vmax[b * NCLS + j] - mx);
    out[tid] = vmax[tid] - mx - logf(s);
  }
}

// ---------------- host orchestration ----------------
extern "C" void kernel_launch(void* const* d_in, const int* in_sizes, int n_in,
                              void* d_out, int out_size, void* d_ws, size_t ws_size,
                              hipStream_t stream) {
  const float* x     = (const float*)d_in[0];
  const float* w1    = (const float*)d_in[1];
  const float* b1    = (const float*)d_in[2];
  const float* w2    = (const float*)d_in[3];
  const float* b2    = (const float*)d_in[4];
  const float* w3    = (const float*)d_in[5];
  const float* b3    = (const float*)d_in[6];
  const float* w_in  = (const float*)d_in[7];
  const float* w_rec = (const float*)d_in[8];
  const float* w_fc  = (const float*)d_in[9];
  const float* b_fc  = (const float*)d_in[10];
  (void)in_sizes; (void)n_in; (void)out_size; (void)ws_size;

  char* ws = (char*)d_ws;
  // workspace layout (all 256B aligned); act2p/feats reuse the dead act1 region
  const size_t act1_off   = 0;
  const size_t act1_sz    = (size_t)BT * 900 * 32 * 2;      // 117,964,800
  const size_t act2_off   = act1_off + act1_sz;
  const size_t act2_sz    = (size_t)BT * 784 * 64 * 2;      // 205,520,896
  const size_t act2p_off  = 0;                               // reuse act1
  const size_t act2p_sz   = (size_t)BT * 196 * 64 * 2;      // 51,380,224
  const size_t feats_off  = act2p_off + act2p_sz;            // fits in act1 region
  const size_t ff_off     = act2_off + act2_sz;
  const size_t ff_sz      = (size_t)BT * HID * 4;
  const size_t pack2_off  = ff_off + ff_sz;
  const size_t pack3_off  = pack2_off + 9 * 4 * 32 * 16 * 2;
  const size_t packin_off = pack3_off + 18 * 6 * 32 * 16 * 2;
  const size_t packrec_off= packin_off + (size_t)432 * 8 * 32 * 16 * 2;

  half_t* act1    = (half_t*)(ws + act1_off);
  half_t* act2    = (half_t*)(ws + act2_off);
  half_t* act2p   = (half_t*)(ws + act2p_off);
  half_t* feats   = (half_t*)(ws + feats_off);
  float*  ff      = (float*)(ws + ff_off);
  half_t* pack2   = (half_t*)(ws + pack2_off);
  half_t* pack3   = (half_t*)(ws + pack3_off);
  half_t* packin  = (half_t*)(ws + packin_off);
  half_t* packrec = (half_t*)(ws + packrec_off);

  // 1. pack weights into WMMA B-fragment order
  pack_w2_k  <<<72,    256, 0, stream>>>(w2, pack2);
  pack_w3_k  <<<216,   256, 0, stream>>>(w3, pack3);
  pack_win_k <<<6912,  256, 0, stream>>>(w_in, packin);
  pack_wrec_k<<<64,    256, 0, stream>>>(w_rec, packrec);

  // 2. encoder
  conv1_k    <<<230400, 256, 0, stream>>>(x, w1, b1, act1);
  conv2_wmma <<<BT * 49, 128, 0, stream>>>(act1, pack2, b2, act2);
  maxpool_k  <<<100352, 256, 0, stream>>>(act2, act2p);
  conv3_wmma <<<BT * 9, 192, 0, stream>>>(act2p, pack3, b3, feats);

  // 3. feed-forward currents
  ff_gemm    <<<128, 256, 0, stream>>>(feats, packin, ff);

  // 4. recurrent LIF scan + readout + log_softmax
  snn_recurrent<<<1, 512, 0, stream>>>(ff, packrec, w_fc, b_fc, (float*)d_out);
}
